// myModel_67070209295036
// MI455X (gfx1250) — matrix operand order repaired and checked by reading, not compiled, not run on hardware
//
#include <hip/hip_runtime.h>
#include <hip/hip_bf16.h>
#include <math.h>

#define BB 4
#define CC 64
#define HH 96
#define WW 320
#define DD 8
#define TILE_W 32
#define XPAD 34      // TILE_W + 2 (3x3 SAME halo)
#define DBATCH 4     // planes per LDS batch

typedef __attribute__((ext_vector_type(16))) __bf16 v16bf;
typedef __attribute__((ext_vector_type(8)))  float  v8f;

struct ShiftTab { int iofs[DD]; float frac[DD]; };

__device__ __forceinline__ unsigned short f2bf(float f) {
  union { float f; unsigned u; } v; v.f = f;
  unsigned u = v.u;
  u += 0x7FFFu + ((u >> 16) & 1u);   // round-to-nearest-even
  return (unsigned short)(u >> 16);
}

union FragU { uint4 q[2]; v16bf v; };

// ---- NCHW f32 -> NHWC f32 (ci contiguous) ----
__global__ void pack_nchw_to_nhwc(const float* __restrict__ in, float* __restrict__ out) {
  int idx = blockIdx.x * blockDim.x + threadIdx.x;
  if (idx >= BB * CC * HH * WW) return;
  int c = idx & 63;
  int w = (idx >> 6) % WW;
  int t = idx / (64 * WW);
  int h = t % HH;
  int b = t / HH;
  out[idx] = in[((b * CC + c) * HH + h) * WW + w];
}

// ---- weights [Co,Ci,3,3] f32 -> bf16 A-fragments [tap][chunk][co_blk][lane][e] ----
__global__ void pack_weights(const float* __restrict__ Wsrc, unsigned short* __restrict__ frag) {
  int idx = blockIdx.x * blockDim.x + threadIdx.x;   // 9*2*4*32*16
  if (idx >= 9 * 2 * 4 * 32 * 16) return;
  int e     = idx & 15;
  int lane  = (idx >> 4) & 31;
  int cob   = (idx >> 9) & 3;
  int chunk = (idx >> 11) & 1;
  int tap   = idx >> 12;
  int ky = tap / 3, kx = tap % 3;
  int co = cob * 16 + (lane & 15);
  // 16-bit A 16x32 layout: K = e + 8*((e>=8) + (lane>=16))
  int k  = e + 8 * ((e >> 3) + (lane >> 4));
  int ci = chunk * 32 + k;
  frag[idx] = f2bf(Wsrc[((co * CC + ci) * 3 + ky) * 3 + kx]);
}

// ---- fused warp + 3x3 conv (implicit GEMM, bf16 WMMA) + max over planes + bias + relu ----
__launch_bounds__(256)
__global__ void plane_conv_max(const float* __restrict__ xin,           // NHWC f32
                               const unsigned short* __restrict__ wfrag, // bf16 A-fragments
                               const float* __restrict__ bias,
                               float* __restrict__ out,
                               ShiftTab st, int nchw_out) {
  __shared__ unsigned int lds_u[DBATCH * 3 * XPAD * CC / 2];   // 52,224 B of bf16
  const int tid  = threadIdx.x;
  const int lane = tid & 31;
  const int wave = tid >> 5;
  const int cob  = wave & 3;          // co block (16 rows)
  const int nb   = wave >> 2;         // spatial block (16 cols)
  const int wt   = blockIdx.x;
  const int h    = blockIdx.y;
  const int b    = blockIdx.z;
  const int w0   = wt * TILE_W;

  v8f mx;
  #pragma unroll
  for (int i = 0; i < 8; ++i) mx[i] = -__builtin_inff();

  for (int batch = 0; batch < 2; ++batch) {
    if (batch) __syncthreads();
    // ---- build warped bf16 tile in LDS: [dl][r][x][ci], 4 planes ----
    for (int idx = tid; idx < DBATCH * 3 * XPAD * 16; idx += 256) {
      int ci4 = idx & 15;
      int x   = (idx >> 4) % XPAD;
      int r   = (idx / (16 * XPAD)) % 3;
      int dl  = idx / (16 * XPAD * 3);
      int d   = batch * DBATCH + dl;
      int yin = h + r - 1;
      int wv  = w0 + x - 1;
      float4 val = make_float4(0.f, 0.f, 0.f, 0.f);
      if ((unsigned)yin < (unsigned)HH && (unsigned)wv < (unsigned)WW) {
        int   y0 = yin + st.iofs[d];
        float f  = st.frac[d];
        float4 g0 = make_float4(0.f, 0.f, 0.f, 0.f);
        float4 g1 = g0;
        if ((unsigned)y0 < (unsigned)HH)
          g0 = *(const float4*)(xin + (((size_t)b * HH + y0) * WW + wv) * CC + ci4 * 4);
        if ((unsigned)(y0 + 1) < (unsigned)HH)
          g1 = *(const float4*)(xin + (((size_t)b * HH + y0 + 1) * WW + wv) * CC + ci4 * 4);
        float c0 = 1.0f - f, c1 = f;
        val.x = c0 * g0.x + c1 * g1.x;
        val.y = c0 * g0.y + c1 * g1.y;
        val.z = c0 * g0.z + c1 * g1.z;
        val.w = c0 * g0.w + c1 * g1.w;
      }
      uint2 pk;
      pk.x = (unsigned)f2bf(val.x) | ((unsigned)f2bf(val.y) << 16);
      pk.y = (unsigned)f2bf(val.z) | ((unsigned)f2bf(val.w) << 16);
      int haddr = ((((dl * 3 + r) * XPAD + x) * CC) + ci4 * 4) >> 1;  // uint index
      *(uint2*)&lds_u[haddr] = pk;
    }
    __syncthreads();

    // ---- implicit GEMM: K = 9 taps x 2 chunks of 32 ci ----
    v8f acc[DBATCH];
    #pragma unroll
    for (int dl = 0; dl < DBATCH; ++dl)
      #pragma unroll
      for (int i = 0; i < 8; ++i) acc[dl][i] = 0.f;

    const unsigned short* ldsh = (const unsigned short*)lds_u;
    #pragma unroll
    for (int tap = 0; tap < 9; ++tap) {
      const int ky = tap / 3, kx = tap % 3;
      #pragma unroll
      for (int chunk = 0; chunk < 2; ++chunk) {
        FragU A;
        const uint4* ap = (const uint4*)(wfrag + ((((tap * 2 + chunk) * 4 + cob) * 32 + lane) * 16));
        A.q[0] = ap[0];
        A.q[1] = ap[1];
        const int x = nb * 16 + (lane & 15) + kx;
        #pragma unroll
        for (int dl = 0; dl < DBATCH; ++dl) {
          FragU Bf;
          int hoff = ((dl * 3 + ky) * XPAD + x) * CC + chunk * 32 + ((lane >> 4) << 3);
          const uint4* bp = (const uint4*)(ldsh + hoff);
          Bf.q[0] = bp[0];
          Bf.q[1] = bp[2];
          acc[dl] = __builtin_amdgcn_wmma_f32_16x16x32_bf16(
              false, A.v, false, Bf.v, (short)0, acc[dl], false, false);
        }
      }
    }
    #pragma unroll
    for (int dl = 0; dl < DBATCH; ++dl)
      #pragma unroll
      for (int i = 0; i < 8; ++i) mx[i] = fmaxf(mx[i], acc[dl][i]);
  }

  // ---- epilogue: max already folded; + bias, relu, store ----
  const int n   = w0 + nb * 16 + (lane & 15);        // output w
  const int coB = cob * 16 + ((lane >> 4) << 3);     // D layout: M = v + 8*(lane>=16)
  #pragma unroll
  for (int v = 0; v < 8; ++v) {
    int co = coB + v;
    float r = fmaxf(mx[v] + bias[co], 0.0f);
    size_t addr = nchw_out
        ? ((((size_t)b * CC + co) * HH + h) * WW + n)
        : ((((size_t)b * HH + h) * WW + n) * CC + co);
    out[addr] = r;
  }
}

extern "C" void kernel_launch(void* const* d_in, const int* in_sizes, int n_in,
                              void* d_out, int out_size, void* d_ws, size_t ws_size,
                              hipStream_t stream) {
  (void)in_sizes; (void)n_in; (void)out_size; (void)ws_size;
  const float* x  = (const float*)d_in[0];
  const float* W1 = (const float*)d_in[1];
  const float* b1 = (const float*)d_in[2];
  const float* W2 = (const float*)d_in[3];
  const float* b2 = (const float*)d_in[4];
  float* outp = (float*)d_out;

  char* ws = (char*)d_ws;
  const size_t xnBytes = (size_t)BB * HH * WW * CC * sizeof(float);   // 31.5 MB
  float*          XN = (float*)ws;
  float*          Y1 = (float*)(ws + xnBytes);
  unsigned short* F1 = (unsigned short*)(ws + 2 * xnBytes);
  unsigned short* F2 = F1 + 9 * 2 * 4 * 32 * 16;

  // plane shifts: 76 * linspace(1/50, 1, 8); constant over h -> integer offset + frac
  ShiftTab st;
  for (int d = 0; d < DD; ++d) {
    float disp = (1.0f / 50.0f) + (float)d * ((1.0f - 1.0f / 50.0f) / 7.0f);
    float s  = 76.0f * disp;
    float fl = floorf(s);
    st.iofs[d] = (int)fl;
    st.frac[d] = s - fl;
  }

  const int nElem = BB * CC * HH * WW;
  pack_nchw_to_nhwc<<<(nElem + 255) / 256, 256, 0, stream>>>(x, XN);
  pack_weights<<<(9 * 2 * 4 * 32 * 16 + 255) / 256, 256, 0, stream>>>(W1, F1);
  pack_weights<<<(9 * 2 * 4 * 32 * 16 + 255) / 256, 256, 0, stream>>>(W2, F2);

  dim3 grid(WW / TILE_W, HH, BB);
  plane_conv_max<<<grid, 256, 0, stream>>>(XN, F1, b1, Y1, st, 0);
  plane_conv_max<<<grid, 256, 0, stream>>>(Y1, F2, b2, outp, st, 1);
}
